// Encoder_67611375173867
// MI455X (gfx1250) — compile-verified, matrix operand order checked by reading, not run on hardware
//
#include <hip/hip_runtime.h>
#include <stdint.h>

// Problem constants
#define VOCAB 32000
#define EMBED 1024
#define UNITS 1024
#define BATCH 64
#define SEQL  128
#define GATES 4096   // 4*UNITS
#define MROWS (BATCH*SEQL)   // 8192

typedef __attribute__((ext_vector_type(16))) __bf16 bf16x16;
typedef __attribute__((ext_vector_type(8)))  __bf16 bf16x8;
typedef __attribute__((ext_vector_type(8)))  float  f32x8;

__device__ __forceinline__ float sigm(float x) { return 1.0f / (1.0f + __expf(-x)); }
__device__ __forceinline__ float tanh_fast(float x) { return 2.0f * sigm(2.0f * x) - 1.0f; }

// LDS byte offset of a __shared__ object: generic AS pointer to LDS carries the
// LDS offset in its low 32 bits (aperture in the high bits).
__device__ __forceinline__ uint32_t lds_off(const void* p) {
  return (uint32_t)(uintptr_t)p;
}

// ---------------------------------------------------------------------------
// Pack a f32 [K=1024][N=4096] weight matrix into bf16 WMMA B-fragment order:
// packed[(ntile*32 + kchunk)*32 + lane][16 elems]; element j (pair v=j>>1):
//   k = kchunk*32 + (v<4?0:16) + (lane>>4)*8 + (v&3)*2 + (j&1)
//   n = ntile*16 + (lane&15)
// Matches the ISA 16-bit 16x32 fragment striping; each lane reads 32
// contiguous bytes per K-chunk.
// ---------------------------------------------------------------------------
__global__ __launch_bounds__(256) void pack_w_kernel(const float* __restrict__ W,
                                                     __bf16* __restrict__ out) {
  unsigned idx = blockIdx.x * 256u + threadIdx.x;     // < 4,194,304
  int j     = idx & 15;
  int lane  = (idx >> 4) & 31;
  int kc    = (idx >> 9) & 31;
  int ntile = idx >> 14;                              // 0..255
  int v = j >> 1;
  int k = kc * 32 + ((v < 4) ? 0 : 16) + (lane >> 4) * 8 + (v & 3) * 2 + (j & 1);
  int n = ntile * 16 + (lane & 15);
  out[idx] = (__bf16)W[(size_t)k * GATES + n];
}

// Embedding gather -> bf16 activations [B*T][1024]
__global__ __launch_bounds__(256) void embed_kernel(const int* __restrict__ tok,
                                                    const float* __restrict__ emb,
                                                    __bf16* __restrict__ xb) {
  size_t idx = (size_t)blockIdx.x * 256 + threadIdx.x;   // < 8192*1024
  int e  = (int)(idx & (EMBED - 1));
  int bt = (int)(idx >> 10);
  int token = tok[bt];
  xb[idx] = (__bf16)emb[(size_t)token * EMBED + e];
}

// Zero the recurrent state (h bf16 ping-pong, h f32, c f32)
__global__ __launch_bounds__(256) void init_state_kernel(__bf16* hb0, __bf16* hb1,
                                                         float* hf, float* cf) {
  int i = blockIdx.x * 256 + threadIdx.x;   // < 65536
  hb0[i] = (__bf16)0.0f;
  hb1[i] = (__bf16)0.0f;
  hf[i] = 0.0f;
  cf[i] = 0.0f;
}

__global__ __launch_bounds__(256) void copy_state_kernel(const float* __restrict__ hf,
                                                         const float* __restrict__ cf,
                                                         float* __restrict__ ho,
                                                         float* __restrict__ co) {
  int i = blockIdx.x * 256 + threadIdx.x;   // < 65536
  ho[i] = hf[i];
  co[i] = cf[i];
}

// ---------------------------------------------------------------------------
// Big input projection: out[M=8192][4096] = A(bf16,[M][1024]) @ W(packed) + bias.
// WG = 8 waves sharing one mtile; wave w owns ntile (ngrp*8 + w).
// Operands are staged in LDS with double-buffered GLOBAL_LOAD_ASYNC_TO_LDS_B128
// (ASYNCcnt) so VMEM latency overlaps v_wmma_f32_16x16x32_bf16 issue, and the
// A chunk is fetched once per wave pair (4x per WG) instead of once per wave.
// Every thread issues exactly 3 async-LDS loads per K-chunk (A segment is
// replicated across waves with identical data so per-wave ASYNCcnt is uniform);
// async loads complete in order, so s_wait_asynccnt 0x3 drains the current
// chunk after the next chunk's prefetch is in flight.
// ---------------------------------------------------------------------------
__global__ __launch_bounds__(256) void gemm_xp_kernel(const __bf16* __restrict__ Ab,
                                                      const __bf16* __restrict__ Wp,
                                                      const float* __restrict__ bias,
                                                      float* __restrict__ out) {
  __shared__ __align__(16) __bf16 Abuf[2][16 * 32];    // 1 KB per buffer
  __shared__ __align__(16) __bf16 Bbuf[2][8 * 512];    // 8 KB per buffer

  const int tid  = threadIdx.x;
  const int wave = tid >> 5;
  const int lane = tid & 31;
  const int half = lane >> 4;
  const int l16  = lane & 15;

  const int ngrp  = blockIdx.x & 31;     // 32 groups of 8 ntiles
  const int mtile = blockIdx.x >> 5;     // 0..511
  const int nt0   = ngrp * 8;
  const int ntile = nt0 + wave;

  // A copy: 64 segments of 16B (16 rows x 64B); segment tid&63, 4x replicated.
  const int as_row = (tid & 63) >> 2;
  const int as_seg = tid & 3;
  const __bf16* a_src = Ab + (size_t)(mtile * 16 + as_row) * 1024 + as_seg * 8;
  const uint32_t a_dst = (uint32_t)((as_row * 32 + as_seg * 8) * 2);
  // B copy: thread (b_nt = tid>>5, lane) copies its lane's 32B chunk.
  const int b_nt = tid >> 5;
  const __bf16* b_src = Wp + ((size_t)(nt0 + b_nt) * 32) * 512 + (size_t)lane * 16;
  const uint32_t b_dst = (uint32_t)((b_nt * 512 + lane * 16) * 2);

  const uint32_t abase0 = lds_off(&Abuf[0][0]);
  const uint32_t abase1 = lds_off(&Abuf[1][0]);
  const uint32_t bbase0 = lds_off(&Bbuf[0][0]);
  const uint32_t bbase1 = lds_off(&Bbuf[1][0]);

  auto prefetch = [&](int kc, int buf) {
    uint64_t asrc = (uint64_t)(uintptr_t)(a_src + kc * 32);
    uint32_t adst = (buf ? abase1 : abase0) + a_dst;
    uint64_t bsrc = (uint64_t)(uintptr_t)(b_src + (size_t)kc * 512);
    uint32_t bdst = (buf ? bbase1 : bbase0) + b_dst;
    asm volatile("global_load_async_to_lds_b128 %0, %1, off"
                 :: "v"(adst), "v"(asrc) : "memory");
    asm volatile("global_load_async_to_lds_b128 %0, %1, off"
                 :: "v"(bdst), "v"(bsrc) : "memory");
    asm volatile("global_load_async_to_lds_b128 %0, %1, off offset:16"
                 :: "v"(bdst), "v"(bsrc) : "memory");
  };

  f32x8 acc = {};
  prefetch(0, 0);
  int buf = 0;
  for (int kc = 0; kc < 32; ++kc) {
    if (kc + 1 < 32) {
      prefetch(kc + 1, buf ^ 1);
      asm volatile("s_wait_asynccnt 0x3" ::: "memory");
    } else {
      asm volatile("s_wait_asynccnt 0x0" ::: "memory");
    }
    __syncthreads();   // make other waves' async-LDS writes visible

    const __bf16* Ash = &Abuf[buf][0];
    const __bf16* Bsh = &Bbuf[buf][0];
    bf16x8 a0 = *(const bf16x8*)(Ash + l16 * 32 + half * 8);
    bf16x8 a1 = *(const bf16x8*)(Ash + l16 * 32 + half * 8 + 16);
    bf16x16 a = __builtin_shufflevector(a0, a1, 0,1,2,3,4,5,6,7,8,9,10,11,12,13,14,15);
    bf16x16 b = *(const bf16x16*)(Bsh + wave * 512 + lane * 16);
    acc = __builtin_amdgcn_wmma_f32_16x16x32_bf16(false, a, false, b,
                                                  (short)0, acc, false, false);
    __syncthreads();   // protect buffer reuse (WAR vs next prefetch)
    buf ^= 1;
  }

  int n = ntile * 16 + l16;
  float bn = bias[n];
#pragma unroll
  for (int r = 0; r < 8; ++r) {
    int row = mtile * 16 + r + half * 8;   // C/D layout: lanes16-31 hold M=8..15
    out[(size_t)row * GATES + n] = acc[r] + bn;
  }
}

// ---------------------------------------------------------------------------
// One LSTM timestep. Grid (64 utiles, 4 mtiles), 128 threads = 4 waves.
// Wave g computes the 16x16 tile of gate g: z = h @ Wh[:, g*1024+utile*16 ..],
// then the WG fuses the i/f/g/o combine + masking through LDS.
// Weight panel streams from L2 (fits easily in 192MB); direct reg loads keep
// the latency-critical recurrence simple.
// ---------------------------------------------------------------------------
__global__ __launch_bounds__(128) void lstm_step_kernel(
    const __bf16* __restrict__ Whp,
    const float*  __restrict__ xp,        // [B*T][4096], row = b*SEQL + t
    const __bf16* __restrict__ h_prev_bf, // [64][1024]
    __bf16*       __restrict__ h_next_bf, // [64][1024]
    const int*    __restrict__ tokens,    // [B][T]
    int t,
    float* __restrict__ cf,               // [64][1024] in/out
    float* __restrict__ hf,               // [64][1024] in/out (f32 carried h)
    float* __restrict__ seq_f32,          // nullable, [(b*T+t)*1024+u]
    __bf16* __restrict__ seq_bf) {        // nullable
  __shared__ float zt[4][16][16];

  const int utile = blockIdx.x;      // 0..63
  const int mtile = blockIdx.y;      // 0..3
  const int tid   = threadIdx.x;
  const int wave  = tid >> 5;        // gate: 0=i 1=f 2=g 3=o
  const int lane  = tid & 31;
  const int half  = lane >> 4;
  const int l16   = lane & 15;

  const int ntile = wave * 64 + utile;
  const __bf16* aptr = h_prev_bf + (size_t)(mtile * 16 + l16) * UNITS + half * 8;
  const __bf16* bptr = Whp + ((size_t)ntile * 32 * 32 + lane) * 16;

  f32x8 acc = {};
#pragma unroll 4
  for (int kc = 0; kc < 32; ++kc) {
    bf16x8 a0 = *(const bf16x8*)(aptr + kc * 32);
    bf16x8 a1 = *(const bf16x8*)(aptr + kc * 32 + 16);
    bf16x16 a = __builtin_shufflevector(a0, a1, 0,1,2,3,4,5,6,7,8,9,10,11,12,13,14,15);
    bf16x16 b = *(const bf16x16*)(bptr + (size_t)kc * 512);
    acc = __builtin_amdgcn_wmma_f32_16x16x32_bf16(false, a, false, b,
                                                  (short)0, acc, false, false);
  }

  // epilogue: z = h@Wh + xp_t  -> LDS
  {
    const int col = wave * UNITS + utile * 16 + l16;
#pragma unroll
    for (int r = 0; r < 8; ++r) {
      int m = r + half * 8;
      int row = mtile * 16 + m;   // batch index
      float z = acc[r] + xp[((size_t)row * SEQL + t) * GATES + col];
      zt[wave][m][l16] = z;
    }
  }
  __syncthreads();

  // fused gate combine + zero-mask semantics; 256 elems over 128 threads
  for (int e = tid; e < 256; e += 128) {
    int r = e >> 4, n = e & 15;
    int b = mtile * 16 + r;
    int u = utile * 16 + n;
    float ig = sigm(zt[0][r][n]);
    float fg = sigm(zt[1][r][n]);
    float gg = tanh_fast(zt[2][r][n]);
    float og = sigm(zt[3][r][n]);
    size_t su = (size_t)b * UNITS + u;
    float c_old = cf[su];
    float c_new = fg * c_old + ig * gg;
    float h_new = og * tanh_fast(c_new);
    bool m = tokens[b * SEQL + t] != 0;
    float h2 = m ? h_new : hf[su];
    float c2 = m ? c_new : c_old;
    cf[su] = c2;
    hf[su] = h2;
    h_next_bf[su] = (__bf16)h2;
    size_t so = ((size_t)b * SEQL + t) * UNITS + u;
    if (seq_f32) seq_f32[so] = h2;
    if (seq_bf)  seq_bf[so]  = (__bf16)h2;
  }
}

// ---------------------------------------------------------------------------
extern "C" void kernel_launch(void* const* d_in, const int* in_sizes, int n_in,
                              void* d_out, int out_size, void* d_ws, size_t ws_size,
                              hipStream_t stream) {
  (void)in_sizes; (void)n_in; (void)out_size; (void)ws_size;
  const int*   tok = (const int*)d_in[0];
  const float* emb = (const float*)d_in[1];
  const float* Wx0 = (const float*)d_in[2];
  const float* Wh0 = (const float*)d_in[3];
  const float* b0  = (const float*)d_in[4];
  const float* Wx1 = (const float*)d_in[5];
  const float* Wh1 = (const float*)d_in[6];
  const float* b1  = (const float*)d_in[7];
  float* out = (float*)d_out;

  // output regions: seq[64*128*1024], h0, c0, h1, c1 (64*1024 each)
  float* out_seq = out;
  float* out_h0  = out + (size_t)BATCH * SEQL * UNITS;
  float* out_c0  = out_h0 + (size_t)BATCH * UNITS;
  float* out_h1  = out_c0 + (size_t)BATCH * UNITS;
  float* out_c1  = out_h1 + (size_t)BATCH * UNITS;

  // workspace carve-up (~200 MB)
  char* ws = (char*)d_ws;
  size_t off = 0;
  auto take = [&](size_t bytes) -> char* {
    char* p = ws + off;
    off += (bytes + 255) & ~(size_t)255;
    return p;
  };
  __bf16* Wx0p  = (__bf16*)take((size_t)1024 * GATES * 2);
  __bf16* Wh0p  = (__bf16*)take((size_t)1024 * GATES * 2);
  __bf16* Wx1p  = (__bf16*)take((size_t)1024 * GATES * 2);
  __bf16* Wh1p  = (__bf16*)take((size_t)1024 * GATES * 2);
  __bf16* x0b   = (__bf16*)take((size_t)MROWS * EMBED * 2);
  __bf16* seq0b = (__bf16*)take((size_t)MROWS * UNITS * 2);
  float*  xp    = (float*)take((size_t)MROWS * GATES * 4);   // reused by both layers
  __bf16* hbf0  = (__bf16*)take((size_t)BATCH * UNITS * 2);
  __bf16* hbf1  = (__bf16*)take((size_t)BATCH * UNITS * 2);
  float*  hf    = (float*)take((size_t)BATCH * UNITS * 4);
  float*  cf    = (float*)take((size_t)BATCH * UNITS * 4);

  const int packBlocks  = (1024 * GATES) / 256;        // 16384
  const int embedBlocks = (MROWS * EMBED) / 256;       // 32768
  const int stateBlocks = (BATCH * UNITS) / 256;       // 256
  const int gemmBlocks  = (MROWS / 16) * 32;           // 512 mtiles * 32 ngroups

  // repack weights to bf16 fragment order (every call; deterministic)
  pack_w_kernel<<<packBlocks, 256, 0, stream>>>(Wx0, Wx0p);
  pack_w_kernel<<<packBlocks, 256, 0, stream>>>(Wh0, Wh0p);
  pack_w_kernel<<<packBlocks, 256, 0, stream>>>(Wx1, Wx1p);
  pack_w_kernel<<<packBlocks, 256, 0, stream>>>(Wh1, Wh1p);

  // embedding gather
  embed_kernel<<<embedBlocks, 256, 0, stream>>>(tok, emb, x0b);

  // ---- layer 0 ----
  init_state_kernel<<<stateBlocks, 256, 0, stream>>>(hbf0, hbf1, hf, cf);
  gemm_xp_kernel<<<gemmBlocks, 256, 0, stream>>>(x0b, Wx0p, b0, xp);
  for (int t = 0; t < SEQL; ++t) {
    lstm_step_kernel<<<dim3(64, 4), 128, 0, stream>>>(
        Wh0p, xp, (t & 1) ? hbf1 : hbf0, (t & 1) ? hbf0 : hbf1,
        tok, t, cf, hf, nullptr, seq0b);
  }
  copy_state_kernel<<<stateBlocks, 256, 0, stream>>>(hf, cf, out_h0, out_c0);

  // ---- layer 1 ----
  init_state_kernel<<<stateBlocks, 256, 0, stream>>>(hbf0, hbf1, hf, cf);
  gemm_xp_kernel<<<gemmBlocks, 256, 0, stream>>>(seq0b, Wx1p, b1, xp);
  for (int t = 0; t < SEQL; ++t) {
    lstm_step_kernel<<<dim3(64, 4), 128, 0, stream>>>(
        Wh1p, xp, (t & 1) ? hbf1 : hbf0, (t & 1) ? hbf0 : hbf1,
        tok, t, cf, hf, out_seq, nullptr);
  }
  copy_state_kernel<<<stateBlocks, 256, 0, stream>>>(hf, cf, out_h1, out_c1);
}
